// FusedLinearForPPO_47785806135378
// MI455X (gfx1250) — compile-verified, matrix operand order checked by reading, not run on hardware
//
#include <hip/hip_runtime.h>
#include <hip/hip_bf16.h>

// ---------------------------------------------------------------------------
// Problem constants (B=4, S=1024, H=2048, V=32000)
// ---------------------------------------------------------------------------
#define HDIM   2048
#define VDIM   32000
#define NROWS  4096            // B*S
#define M_TILE 32              // rows per workgroup
#define N_PER_ITER 256         // columns per N-iteration (8 waves * 32)
#define NSPLIT 5               // vocab splits -> 128*5 = 640 workgroups
#define N_ITERS 25             // 32000 / (256*5)
#define NCOLS_PER_SPLIT 6400
#define LDA_PAD 4112           // 2048*2 bytes + 16B pad -> conflict-free b128
#define NEG_BIG (-3.4e38f)

typedef __attribute__((ext_vector_type(16))) __bf16 v16bf;
typedef __attribute__((ext_vector_type(8)))  float  v8f;

union ABu { uint4 u[2]; v16bf v; };

// WGP-scope prefetch (scope field omitted == SCOPE 0 == pull into all cache
// levels, stop on WGP$ hit). __builtin_prefetch only reaches SE/SYS scope,
// which skips the WGP cache entirely — useless when the data is already L2
// resident. Inline asm gives us the L2 -> L0 stream-ahead we actually want.
__device__ __forceinline__ void prefetch_wgp(const void* p) {
    asm volatile("global_prefetch_b8 %0, off" :: "v"(p));
}

// ---------------------------------------------------------------------------
// fp32 -> bf16 (round-to-nearest-even), 8 elements / thread
// ---------------------------------------------------------------------------
__device__ __forceinline__ unsigned int pack_bf16(float a, float b) {
    union { float f; unsigned int u; } ua, ub;
    ua.f = a; ub.f = b;
    unsigned int ra = (ua.u + 0x7FFFu + ((ua.u >> 16) & 1u)) >> 16;
    unsigned int rb = (ub.u + 0x7FFFu + ((ub.u >> 16) & 1u)) >> 16;
    return (ra & 0xFFFFu) | (rb << 16);
}

__global__ void cvt_bf16_kernel(const float* __restrict__ src,
                                unsigned short* __restrict__ dst, int n8) {
    int i = blockIdx.x * blockDim.x + threadIdx.x;
    if (i >= n8) return;
    const float4* s = (const float4*)src;
    float4 x0 = s[2 * i], x1 = s[2 * i + 1];
    uint4 o;
    o.x = pack_bf16(x0.x, x0.y);
    o.y = pack_bf16(x0.z, x0.w);
    o.z = pack_bf16(x1.x, x1.y);
    o.w = pack_bf16(x1.z, x1.w);
    ((uint4*)dst)[i] = o;
}

// ---------------------------------------------------------------------------
// online (m, s, t) merge helper: s = sum exp(l-m), t = sum exp(l-m)*l
// ---------------------------------------------------------------------------
__device__ __forceinline__ void merge_stats(float& m, float& s, float& t,
                                            float om, float os, float ot) {
    float m2 = fmaxf(m, om);
    float sa = __expf(m - m2);
    float sb = __expf(om - m2);
    s = s * sa + os * sb;
    t = t * sa + ot * sb;
    m = m2;
}

// ---------------------------------------------------------------------------
// Fused GEMM (bf16 WMMA) + online log-softmax statistics
//   grid = (128 M-blocks, 5 N-splits), block = 256 threads (8 waves)
// ---------------------------------------------------------------------------
__global__ void __launch_bounds__(256)
fused_lm_head_kernel(const unsigned short* __restrict__ Wbf,   // [V][H] bf16
                     const unsigned short* __restrict__ Abf,   // [NROWS][H] bf16
                     const int* __restrict__ ids,              // [NROWS]
                     float4* __restrict__ partial)             // [NROWS*NSPLIT]
{
    extern __shared__ char lds[];                  // A tile + wave stats
    const int tid    = threadIdx.x;
    const int lane   = tid & 31;
    const int wave   = tid >> 5;
    const int laneLo = lane & 15;
    const int laneHi = lane >> 4;                  // 0 or 1 (K half / row half)
    const int mblk   = blockIdx.x;
    const int nspl   = blockIdx.y;
    const int row0   = mblk * M_TILE;

    // ---- stage A tile (32 rows x 4096B) into padded LDS -------------------
    {
        const uint4* gsrc = (const uint4*)(Abf + (size_t)row0 * HDIM);
        #pragma unroll
        for (int i = 0; i < 32; ++i) {
            int c   = tid + i * 256;               // 8192 16B-chunks total
            int r   = c >> 8;                      // 256 chunks per row
            int off = c & 255;
            uint4 v = gsrc[r * 256 + off];
            *(uint4*)(lds + r * LDA_PAD + off * 16) = v;
        }
    }
    __syncthreads();

    // ---- per-lane online stats for the 16 rows this lane touches ----------
    // q = msub*8 + j  ->  local row = msub*16 + j + laneHi*8
    float mx[16], ss[16], tt[16], gg[16];
    int   tgt[16];
    #pragma unroll
    for (int q = 0; q < 16; ++q) {
        mx[q] = NEG_BIG; ss[q] = 0.f; tt[q] = 0.f; gg[q] = 0.f;
        int msub = q >> 3, j = q & 7;
        tgt[q] = ids[row0 + msub * 16 + j + laneHi * 8];
    }

    // per-lane LDS bases for the two A sub-tiles (WMMA A layout)
    const char* ldsA0 = lds + (0 * 16 + laneLo) * LDA_PAD + laneHi * 16;
    const char* ldsA1 = lds + (1 * 16 + laneLo) * LDA_PAD + laneHi * 16;

    for (int it = 0; it < N_ITERS; ++it) {
        const int nbase = nspl * NCOLS_PER_SPLIT + it * N_PER_ITER + wave * 32;
        // B operand: lane = vocab row (column of logits), row-major W -> no transpose
        const char* bptr0 = (const char*)Wbf + (size_t)(nbase + laneLo) * (HDIM * 2)
                            + laneHi * 16;
        const char* bptr1 = bptr0 + 16 * (HDIM * 2);

        v8f acc[2][2] = {};                        // [msub][nsub] 16x16 f32 tiles

        #pragma unroll 8
        for (int kk = 0; kk < 64; ++kk) {
            const int kb = kk * 64;
            if ((kk & 7) == 0) {                   // WGP-scope B stream-ahead
                prefetch_wgp(bptr0 + kb + 1024);
                prefetch_wgp(bptr1 + kb + 1024);
            }
            ABu a0, a1, b0, b1;
            a0.u[0] = *(const uint4*)(ldsA0 + kb);
            a0.u[1] = *(const uint4*)(ldsA0 + kb + 32);
            a1.u[0] = *(const uint4*)(ldsA1 + kb);
            a1.u[1] = *(const uint4*)(ldsA1 + kb + 32);
            b0.u[0] = *(const uint4*)(bptr0 + kb);
            b0.u[1] = *(const uint4*)(bptr0 + kb + 32);
            b1.u[0] = *(const uint4*)(bptr1 + kb);
            b1.u[1] = *(const uint4*)(bptr1 + kb + 32);

            acc[0][0] = __builtin_amdgcn_wmma_f32_16x16x32_bf16(
                false, a0.v, false, b0.v, (short)0, acc[0][0], false, false);
            acc[1][0] = __builtin_amdgcn_wmma_f32_16x16x32_bf16(
                false, a1.v, false, b0.v, (short)0, acc[1][0], false, false);
            acc[0][1] = __builtin_amdgcn_wmma_f32_16x16x32_bf16(
                false, a0.v, false, b1.v, (short)0, acc[0][1], false, false);
            acc[1][1] = __builtin_amdgcn_wmma_f32_16x16x32_bf16(
                false, a1.v, false, b1.v, (short)0, acc[1][1], false, false);
        }

        // ---- fold this 32x(2x16) logits tile into online stats ------------
        const int col0 = nbase + laneLo;           // nsub = 0 column
        const int col1 = nbase + 16 + laneLo;      // nsub = 1 column
        #pragma unroll
        for (int msub = 0; msub < 2; ++msub) {
            #pragma unroll
            for (int j = 0; j < 8; ++j) {
                const int q = msub * 8 + j;
                float v0 = acc[msub][0][j];
                float v1 = acc[msub][1][j];
                gg[q] += (tgt[q] == col0) ? v0 : 0.f;   // target-logit gather
                gg[q] += (tgt[q] == col1) ? v1 : 0.f;
                float m2 = fmaxf(mx[q], fmaxf(v0, v1));
                float sc = __expf(mx[q] - m2);          // 0 on first touch
                float e0 = __expf(v0 - m2);
                float e1 = __expf(v1 - m2);
                ss[q] = ss[q] * sc + e0 + e1;
                tt[q] = tt[q] * sc + e0 * v0 + e1 * v1;
                mx[q] = m2;
            }
        }
    }

    // ---- reduce across the 16 lanes sharing each row (stay within halves) --
    #pragma unroll
    for (int q = 0; q < 16; ++q) {
        #pragma unroll
        for (int d = 1; d < 16; d <<= 1) {
            float om = __shfl_xor(mx[q], d, 32);
            float os = __shfl_xor(ss[q], d, 32);
            float ot = __shfl_xor(tt[q], d, 32);
            float og = __shfl_xor(gg[q], d, 32);
            merge_stats(mx[q], ss[q], tt[q], om, os, ot);
            gg[q] += og;
        }
    }

    // ---- cross-wave merge via LDS ------------------------------------------
    float4* stats = (float4*)(lds + M_TILE * LDA_PAD);    // [8 waves][32 rows]
    __syncthreads();                                       // A tile no longer needed
    if (laneLo == 0) {
        #pragma unroll
        for (int q = 0; q < 16; ++q) {
            int msub = q >> 3, j = q & 7;
            int rloc = msub * 16 + j + laneHi * 8;
            stats[wave * M_TILE + rloc] = make_float4(mx[q], ss[q], tt[q], gg[q]);
        }
    }
    __syncthreads();
    if (tid < M_TILE) {
        float m = NEG_BIG, s = 0.f, t = 0.f, g = 0.f;
        #pragma unroll
        for (int w = 0; w < 8; ++w) {
            float4 p = stats[w * M_TILE + tid];
            merge_stats(m, s, t, p.x, p.y, p.z);
            g += p.w;
        }
        partial[(size_t)(row0 + tid) * NSPLIT + nspl] = make_float4(m, s, t, g);
    }
}

// ---------------------------------------------------------------------------
// Combine the NSPLIT partials -> token_log_prob, entropy
// ---------------------------------------------------------------------------
__global__ void merge_kernel(const float4* __restrict__ partial,
                             float* __restrict__ out) {
    int r = blockIdx.x * blockDim.x + threadIdx.x;
    if (r >= NROWS) return;
    float m = NEG_BIG, s = 0.f, t = 0.f, g = 0.f;
    #pragma unroll
    for (int p = 0; p < NSPLIT; ++p) {
        float4 q = partial[(size_t)r * NSPLIT + p];
        merge_stats(m, s, t, q.x, q.y, q.z);
        g += q.w;
    }
    float lse = m + __logf(s);
    out[r]         = g - lse;          // token log-prob
    out[NROWS + r] = lse - t / s;      // entropy = lse - E[logit]
}

// ---------------------------------------------------------------------------
extern "C" void kernel_launch(void* const* d_in, const int* in_sizes, int n_in,
                              void* d_out, int out_size, void* d_ws, size_t ws_size,
                              hipStream_t stream) {
    const float* hidden  = (const float*)d_in[0];   // [4,1024,2048] fp32
    const float* weights = (const float*)d_in[1];   // [32000,2048] fp32
    const int*   ids     = (const int*)d_in[2];     // [4,1024]
    float*       out     = (float*)d_out;           // 2 * 4096 fp32

    char* ws = (char*)d_ws;
    const size_t wbytes = (size_t)VDIM * HDIM * 2;       // 131,072,000
    const size_t abytes = (size_t)NROWS * HDIM * 2;      //  16,777,216
    unsigned short* Wbf = (unsigned short*)ws;
    unsigned short* Abf = (unsigned short*)(ws + wbytes);
    float4* partial     = (float4*)(ws + wbytes + abytes);

    const int nW8 = VDIM * HDIM / 8;    // 8,192,000
    const int nA8 = NROWS * HDIM / 8;   // 1,048,576
    cvt_bf16_kernel<<<(nW8 + 255) / 256, 256, 0, stream>>>(weights, Wbf, nW8);
    cvt_bf16_kernel<<<(nA8 + 255) / 256, 256, 0, stream>>>(hidden, Abf, nA8);

    const size_t shmem = (size_t)M_TILE * LDA_PAD + 8 * M_TILE * sizeof(float4);
    hipFuncSetAttribute((const void*)fused_lm_head_kernel,
                        hipFuncAttributeMaxDynamicSharedMemorySize, (int)shmem);
    fused_lm_head_kernel<<<dim3(NROWS / M_TILE, NSPLIT), 256, shmem, stream>>>(
        Wbf, Abf, ids, partial);

    merge_kernel<<<(NROWS + 255) / 256, 256, 0, stream>>>(partial, out);
}
